// ResnetPointnet_63788854280434
// MI455X (gfx1250) — matrix operand order repaired and checked by reading, not compiled, not run on hardware
//
#include <hip/hip_runtime.h>

// ---------------------------------------------------------------------------
// ResnetPointnet + plane-scatter + UNet for MI455X (gfx1250), wave32 + WMMA.
// All heavy math runs through v_wmma_f32_16x16x32_bf16 (bf16 in, f32 acc).
// ---------------------------------------------------------------------------

typedef __attribute__((ext_vector_type(16))) __bf16 v16bf;
typedef __attribute__((ext_vector_type(8)))  float  v8f;

union FragU { v16bf v; uint4 q[2]; };

#define WMMA_BF16(A, B, C) \
  __builtin_amdgcn_wmma_f32_16x16x32_bf16(false, (A), false, (B), (short)0, (C), false, false)

// ---------------------------------------------------------------------------
// Encoder GEMM:  out = act(A) @ W^T + bias [+ add]
//   A  : [M,K]  bf16 row-major
//   W  : [N,K]  bf16 row-major  (so lane-n B fragment = contiguous K run)
//   out: bf16 and/or f32 [M,N]
// One wave per 16 rows; loops N in chunks of 64 (4 wmma tiles, A reused).
// ---------------------------------------------------------------------------
template <bool RELU_A>
__global__ void __launch_bounds__(128)
gemm_bf16_wmma(const __bf16* __restrict__ A, const __bf16* __restrict__ W,
               const float* __restrict__ bias, const float* __restrict__ add,
               __bf16* __restrict__ outBf, float* __restrict__ outF,
               int M, int N, int K) {
  const int lane = threadIdx.x & 31;
  const int wv   = threadIdx.x >> 5;
  const int lr = lane & 15, lh = lane >> 4;
  const int m0 = (blockIdx.x * 4 + wv) * 16;
  if (m0 >= M) return;  // wave-uniform

  for (int n0c = 0; n0c < N; n0c += 64) {
    v8f acc[4];
#pragma unroll
    for (int t = 0; t < 4; ++t)
#pragma unroll
      for (int i = 0; i < 8; ++i) acc[t][i] = 0.f;

    for (int kb = 0; kb < K; kb += 32) {
      // A fragment: lanes 0-15 rows m0+lr with K kb+{0..7,16..23};
      //             lanes 16-31 same rows, K kb+{8..15,24..31}.
      FragU fa;
      const __bf16* ap = A + (size_t)(m0 + lr) * K + kb + lh * 8;
      fa.q[0] = *(const uint4*)ap;
      fa.q[1] = *(const uint4*)(ap + 16);
      if (RELU_A) {
#pragma unroll
        for (int i = 0; i < 16; ++i) {
          float f = (float)fa.v[i];
          fa.v[i] = (__bf16)(f > 0.f ? f : 0.f);
        }
      }
#pragma unroll
      for (int t = 0; t < 4; ++t) {
        // B fragment: lane column n, half-wave selects K kb+{0..7,16..23} / {8..15,24..31}.
        FragU fb;
        const __bf16* wp = W + (size_t)(n0c + t * 16 + lr) * K + kb + lh * 8;
        fb.q[0] = *(const uint4*)wp;
        fb.q[1] = *(const uint4*)(wp + 16);
        acc[t] = WMMA_BF16(fa.v, fb.v, acc[t]);
      }
    }
    // D layout: VGPR r -> row m0 + lh*8 + r, col n0 + lr
#pragma unroll
    for (int t = 0; t < 4; ++t) {
      const int n = n0c + t * 16 + lr;
      const float bv = bias ? bias[n] : 0.f;
#pragma unroll
      for (int r = 0; r < 8; ++r) {
        const int m = m0 + lh * 8 + r;
        float v = acc[t][r] + bv;
        if (add)   v += add[(size_t)m * N + n];
        if (outF)  outF[(size_t)m * N + n] = v;
        if (outBf) outBf[(size_t)m * N + n] = (__bf16)v;
      }
    }
  }
}

// ---------------------------------------------------------------------------
// NHWC implicit-GEMM conv (KS=3 SAME or KS=1). Weights pre-packed as
// [tap][Cout][Cin] bf16. Wave tile = 16 pixels (along W) x 64 out channels.
// ---------------------------------------------------------------------------
template <int KS>
__global__ void __launch_bounds__(128)
conv_nhwc_wmma(const __bf16* __restrict__ X, const __bf16* __restrict__ Wt,
               const float* __restrict__ bias, __bf16* __restrict__ outBf,
               float* __restrict__ outF,
               int Nimg, int Hh, int Ww, int Cin, int Cout, int doRelu) {
  const int lane = threadIdx.x & 31;
  const int wv   = threadIdx.x >> 5;
  const int lr = lane & 15, lh = lane >> 4;
  const int xtiles = Ww >> 4;
  int wt = blockIdx.x * 4 + wv;
  const int xt = wt % xtiles; wt /= xtiles;
  const int y  = wt % Hh;
  const int n  = wt / Hh;
  if (n >= Nimg) return;
  const int x0 = xt * 16;
  const int c0 = blockIdx.y * 64;

  v8f acc[4];
#pragma unroll
  for (int t = 0; t < 4; ++t)
#pragma unroll
    for (int i = 0; i < 8; ++i) acc[t][i] = 0.f;

  const int R = KS >> 1;
#pragma unroll
  for (int ky = -R; ky <= R; ++ky) {
    const int yy = y + ky;
    const bool yok = ((unsigned)yy < (unsigned)Hh);
#pragma unroll
    for (int kx = -R; kx <= R; ++kx) {
      const int tap = (ky + R) * KS + (kx + R);
      const __bf16* wb = Wt + (size_t)tap * Cout * Cin;
      const int xx = x0 + lr + kx;
      const bool ok = yok && ((unsigned)xx < (unsigned)Ww);
      const __bf16* arow =
          X + ((size_t)(n * Hh + (yok ? yy : 0)) * Ww + (ok ? xx : 0)) * Cin;
      for (int kb = 0; kb < Cin; kb += 32) {
        const uint4 z = make_uint4(0u, 0u, 0u, 0u);
        FragU fa;
        const __bf16* ap = arow + kb + lh * 8;
        fa.q[0] = ok ? *(const uint4*)ap        : z;
        fa.q[1] = ok ? *(const uint4*)(ap + 16) : z;
#pragma unroll
        for (int t = 0; t < 4; ++t) {
          FragU fb;
          const __bf16* wp = wb + (size_t)(c0 + t * 16 + lr) * Cin + kb + lh * 8;
          fb.q[0] = *(const uint4*)wp;
          fb.q[1] = *(const uint4*)(wp + 16);
          acc[t] = WMMA_BF16(fa.v, fb.v, acc[t]);
        }
      }
    }
  }
#pragma unroll
  for (int t = 0; t < 4; ++t) {
    const int c = c0 + t * 16 + lr;
    const float bv = bias[c];
#pragma unroll
    for (int r = 0; r < 8; ++r) {
      const int x = x0 + lh * 8 + r;
      float v = acc[t][r] + bv;
      if (doRelu) v = v > 0.f ? v : 0.f;
      const size_t o = ((size_t)(n * Hh + y) * Ww + x) * Cout + c;
      if (outF) outF[o] = v;
      else      outBf[o] = (__bf16)v;
    }
  }
}

// ---------------------------------------------------------------------------
// 2x2-stride-2 conv-transpose as 4 independent 1x1 GEMMs (grid.z = tap),
// scattering into a concat buffer with channel stride CoutTot.
// ---------------------------------------------------------------------------
__global__ void __launch_bounds__(128)
upconv_nhwc_wmma(const __bf16* __restrict__ X, const __bf16* __restrict__ Wt,
                 __bf16* __restrict__ out,
                 int Nimg, int Hh, int Ww, int Cin, int Cout, int CoutTot) {
  const int lane = threadIdx.x & 31;
  const int wv   = threadIdx.x >> 5;
  const int lr = lane & 15, lh = lane >> 4;
  const int tap = blockIdx.z, dy = tap >> 1, dx = tap & 1;
  const int xtiles = Ww >> 4;
  int wt = blockIdx.x * 4 + wv;
  const int xt = wt % xtiles; wt /= xtiles;
  const int y  = wt % Hh;
  const int n  = wt / Hh;
  if (n >= Nimg) return;
  const int x0 = xt * 16, c0 = blockIdx.y * 64;
  const __bf16* wb = Wt + (size_t)tap * Cout * Cin;

  v8f acc[4];
#pragma unroll
  for (int t = 0; t < 4; ++t)
#pragma unroll
    for (int i = 0; i < 8; ++i) acc[t][i] = 0.f;

  const __bf16* arow = X + ((size_t)(n * Hh + y) * Ww + x0 + lr) * Cin;
  for (int kb = 0; kb < Cin; kb += 32) {
    FragU fa;
    const __bf16* ap = arow + kb + lh * 8;
    fa.q[0] = *(const uint4*)ap;
    fa.q[1] = *(const uint4*)(ap + 16);
#pragma unroll
    for (int t = 0; t < 4; ++t) {
      FragU fb;
      const __bf16* wp = wb + (size_t)(c0 + t * 16 + lr) * Cin + kb + lh * 8;
      fb.q[0] = *(const uint4*)wp;
      fb.q[1] = *(const uint4*)(wp + 16);
      acc[t] = WMMA_BF16(fa.v, fb.v, acc[t]);
    }
  }
  const int H2 = Hh * 2, W2 = Ww * 2;
#pragma unroll
  for (int t = 0; t < 4; ++t) {
    const int c = c0 + t * 16 + lr;
#pragma unroll
    for (int r = 0; r < 8; ++r) {
      const int xi = x0 + lh * 8 + r;
      out[((size_t)(n * H2 + 2 * y + dy) * W2 + 2 * xi + dx) * CoutTot + c] =
          (__bf16)acc[t][r];
    }
  }
}

// -------------------------- elementwise helpers -----------------------------

__global__ void k_cvt_bf16(const float* __restrict__ s, __bf16* __restrict__ d,
                           size_t n) {
  size_t i = (size_t)blockIdx.x * blockDim.x + threadIdx.x;
  if (i < n) d[i] = (__bf16)s[i];
}

// OIHW f32 -> [tap][O][I] bf16
__global__ void k_prep_conv(const float* __restrict__ w, __bf16* __restrict__ d,
                            int O, int I, int KH, int KW) {
  size_t i = (size_t)blockIdx.x * blockDim.x + threadIdx.x;
  size_t total = (size_t)O * I * KH * KW;
  if (i >= total) return;
  int kx = (int)(i % KW); size_t r = i / KW;
  int ky = (int)(r % KH); r /= KH;
  int ii = (int)(r % I);
  int o  = (int)(r / I);
  d[(((size_t)(ky * KW + kx)) * O + o) * I + ii] = (__bf16)w[i];
}

// IOHW (2x2) f32 -> [tap][O][I] bf16
__global__ void k_prep_up(const float* __restrict__ w, __bf16* __restrict__ d,
                          int I, int O) {
  size_t i = (size_t)blockIdx.x * blockDim.x + threadIdx.x;
  size_t total = (size_t)I * O * 4;
  if (i >= total) return;
  int dx = (int)(i & 1); size_t r = i >> 1;
  int dy = (int)(r & 1); r >>= 1;
  int o  = (int)(r % O);
  int ii = (int)(r / O);
  d[(((size_t)(dy * 2 + dx)) * O + o) * I + ii] = (__bf16)w[i];
}

__global__ void k_fcpos(const float* __restrict__ p, const float* __restrict__ w,
                        const float* __restrict__ b, __bf16* __restrict__ X0) {
  size_t i = (size_t)blockIdx.x * 256 + threadIdx.x;  // 32768*256
  int c = (int)(i & 255); size_t row = i >> 8;
  float a = b[c] + p[row * 3 + 0] * w[c * 3 + 0] + p[row * 3 + 1] * w[c * 3 + 1] +
            p[row * 3 + 2] * w[c * 3 + 2];
  X0[i] = (__bf16)a;
}

__global__ void __launch_bounds__(128)
k_poolT(const __bf16* __restrict__ net, __bf16* __restrict__ pooled) {
  const int b = blockIdx.x, c = threadIdx.x;
  float m = -3.0e38f;
  const __bf16* base = net + (size_t)b * 2048 * 128 + c;
  for (int t = 0; t < 2048; ++t) m = fmaxf(m, (float)base[(size_t)t * 128]);
  pooled[b * 128 + c] = (__bf16)m;
}

__global__ void k_buildX(const __bf16* __restrict__ net,
                         const __bf16* __restrict__ pooled,
                         __bf16* __restrict__ X) {
  size_t i = (size_t)blockIdx.x * 256 + threadIdx.x;  // 32768*256
  int c = (int)(i & 255); size_t row = i >> 8; int b = (int)(row >> 11);
  X[i] = (c < 128) ? net[row * 128 + c] : pooled[b * 128 + (c - 128)];
}

__global__ void __launch_bounds__(128)
k_scatter(const __bf16* __restrict__ net, const float* __restrict__ cmat,
          const float* __restrict__ p, float* __restrict__ cgrid,
          float* __restrict__ cnt) {
  const int bt = blockIdx.x;          // b*2048 + t
  const int b  = bt >> 11;
  const int ch = threadIdx.x;         // 128
  const float inv_md = 1.f / 0.55f;
  const float inv_int = 63.f / 2.f;   // 1/INTERVAL
  const float p0 = p[bt * 3 + 0] * inv_md;
  const float p1 = p[bt * 3 + 1] * inv_md;
  const float p2 = p[bt * 3 + 2] * inv_md;
  const float f = (float)net[(size_t)bt * 128 + ch];
  for (int l = 0; l < 4; ++l) {
    const float* C = cmat + l * 12;
    const float inc = 1.f / (C[9] + 0.01f);
    const float pr0 = (p0 * C[0] + p1 * C[1] + p2 * C[2]) * inc;
    const float pr1 = (p0 * C[3] + p1 * C[4] + p2 * C[5]) * inc;
    int ix = (int)rintf((pr0 + 1.f) * inv_int);
    int iy = (int)rintf((pr1 + 1.f) * inv_int);
    ix = ix < 0 ? 0 : (ix > 63 ? 63 : ix);
    iy = iy < 0 ? 0 : (iy > 63 ? 63 : iy);
    const size_t cell = (((size_t)b * 4 + l) * 64 + ix) * 64 + iy;
    atomicAdd(&cgrid[cell * 128 + ch], f);
    if (ch == 0) atomicAdd(&cnt[cell], 1.f);
  }
}

__global__ void k_norm_g0(const float* __restrict__ cg,
                          const float* __restrict__ cnt,
                          __bf16* __restrict__ g0) {
  size_t i = (size_t)blockIdx.x * 256 + threadIdx.x;  // 64*64*64*128
  float c = cnt[i >> 7];
  g0[i] = (__bf16)(cg[i] / fmaxf(c, 1.f));
}

__global__ void k_pool2(const __bf16* __restrict__ X, __bf16* __restrict__ Y,
                        int Nimg, int Hh, int Ww, int C) {
  size_t i = (size_t)blockIdx.x * blockDim.x + threadIdx.x;
  size_t total = (size_t)Nimg * (Hh >> 1) * (Ww >> 1) * C;
  if (i >= total) return;
  int c = (int)(i % C); size_t r = i / C;
  int x = (int)(r % (Ww >> 1)); r /= (Ww >> 1);
  int y = (int)(r % (Hh >> 1));
  int n = (int)(r / (Hh >> 1));
  const __bf16* b = X + ((size_t)(n * Hh + 2 * y) * Ww + 2 * x) * C + c;
  float m = fmaxf(fmaxf((float)b[0], (float)b[C]),
                  fmaxf((float)b[(size_t)Ww * C], (float)b[(size_t)Ww * C + C]));
  Y[i] = (__bf16)m;
}

__global__ void k_copych(const __bf16* __restrict__ src, __bf16* __restrict__ dst,
                         int npix, int Cs, int Cd, int coff) {
  size_t i = (size_t)blockIdx.x * blockDim.x + threadIdx.x;
  size_t total = (size_t)npix * Cs;
  if (i >= total) return;
  int c = (int)(i % Cs); size_t pix = i / Cs;
  dst[pix * Cd + coff + c] = src[i];
}

// C_MAT (4 planes x 4 x 3), double precision, single thread.
__global__ void k_init_cmat(float* __restrict__ cmat, float* __restrict__ tail) {
  if (threadIdx.x != 0 || blockIdx.x != 0) return;
  const double P[4][3] = {{1, 1, 1}, {1, 0, 0}, {0, 1, 0}, {0, 0, 1}};
  for (int l = 0; l < 4; ++l) {
    double res[12];
    double n0 = P[l][0], n1 = P[l][1], n2 = P[l][2];
    double nl = sqrt(n0 * n0 + n1 * n1 + n2 * n2);
    n0 /= nl; n1 /= nl; n2 /= nl;
    if (n0 == 0.0 && n1 == 0.0 && n2 == 1.0) {
      const double I[12] = {1, 0, 0, 0, 1, 0, 0, 0, 1, 1, 1, 1};
      for (int i = 0; i < 12; ++i) res[i] = I[i];
    } else {
      const double v0 = -n1, v1 = n0, v2 = 0.0;
      const double vv = v0 * v0 + v1 * v1 + v2 * v2;
      const double s = (1.0 - n2) / vv;
      double S[3][3] = {{0, -v2, v1}, {v2, 0, -v0}, {-v1, v0, 0}};
      double S2[3][3], Rm[3][3];
      for (int r = 0; r < 3; ++r)
        for (int c = 0; c < 3; ++c) {
          double a = 0;
          for (int k = 0; k < 3; ++k) a += S[r][k] * S[k][c];
          S2[r][c] = a;
        }
      for (int r = 0; r < 3; ++r)
        for (int c = 0; c < 3; ++c)
          Rm[r][c] = (r == c ? 1.0 : 0.0) + S[r][c] + S2[r][c] * s;
      double bx[3] = {fabs(Rm[0][0]), fabs(Rm[1][0]), fabs(Rm[2][0])};
      double by[3] = {fabs(Rm[0][1]), fabs(Rm[1][1]), fabs(Rm[2][1])};
      double cx = (bx[0] + bx[1] + bx[2]) /
                  sqrt(bx[0] * bx[0] + bx[1] * bx[1] + bx[2] * bx[2]);
      double cy = (by[0] + by[1] + by[2]) /
                  sqrt(by[0] * by[0] + by[1] * by[1] + by[2] * by[2]);
      double mc = cx > cy ? cx : cy;
      double det = Rm[0][0] * (Rm[1][1] * Rm[2][2] - Rm[1][2] * Rm[2][1]) -
                   Rm[0][1] * (Rm[1][0] * Rm[2][2] - Rm[1][2] * Rm[2][0]) +
                   Rm[0][2] * (Rm[1][0] * Rm[2][1] - Rm[1][1] * Rm[2][0]);
      res[0] = (Rm[1][1] * Rm[2][2] - Rm[1][2] * Rm[2][1]) / det;
      res[1] = (Rm[0][2] * Rm[2][1] - Rm[0][1] * Rm[2][2]) / det;
      res[2] = (Rm[0][1] * Rm[1][2] - Rm[0][2] * Rm[1][1]) / det;
      res[3] = (Rm[1][2] * Rm[2][0] - Rm[1][0] * Rm[2][2]) / det;
      res[4] = (Rm[0][0] * Rm[2][2] - Rm[0][2] * Rm[2][0]) / det;
      res[5] = (Rm[0][2] * Rm[1][0] - Rm[0][0] * Rm[1][2]) / det;
      res[6] = (Rm[1][0] * Rm[2][1] - Rm[1][1] * Rm[2][0]) / det;
      res[7] = (Rm[0][1] * Rm[2][0] - Rm[0][0] * Rm[2][1]) / det;
      res[8] = (Rm[0][0] * Rm[1][1] - Rm[0][1] * Rm[1][0]) / det;
      res[9] = res[10] = res[11] = mc;
    }
    for (int i = 0; i < 12; ++i) {
      float f = (float)res[i];
      cmat[l * 12 + i] = f;
      tail[l * 12 + i] = f;
    }
  }
}

// ---------------------------------------------------------------------------

extern "C" void kernel_launch(void* const* d_in, const int* in_sizes, int n_in,
                              void* d_out, int out_size, void* d_ws, size_t ws_size,
                              hipStream_t stream) {
  (void)in_sizes; (void)n_in; (void)out_size; (void)ws_size;
  // JAX tree-flatten order (dict keys sorted):
  // 0: p ; blocks[i] -> {b0,b1,w0,w1,ws} at 1+5i ; 26: fc_pos_b ; 27: fc_pos_w ;
  // unet: b1a..b5b,bf (28..38), w1a..w5b,wf,wu1,wu2 (39..51)
  const float* p   = (const float*)d_in[0];
  const float* rb0[5], *rb1[5], *rw0[5], *rw1[5], *rws[5];
  for (int i = 0; i < 5; ++i) {
    rb0[i] = (const float*)d_in[1 + 5 * i + 0];
    rb1[i] = (const float*)d_in[1 + 5 * i + 1];
    rw0[i] = (const float*)d_in[1 + 5 * i + 2];
    rw1[i] = (const float*)d_in[1 + 5 * i + 3];
    rws[i] = (const float*)d_in[1 + 5 * i + 4];
  }
  const float* fcb = (const float*)d_in[26];
  const float* fcw = (const float*)d_in[27];
  const float* b1a = (const float*)d_in[28], *b1b = (const float*)d_in[29];
  const float* b2a = (const float*)d_in[30], *b2b = (const float*)d_in[31];
  const float* b3a = (const float*)d_in[32], *b3b = (const float*)d_in[33];
  const float* b4a = (const float*)d_in[34], *b4b = (const float*)d_in[35];
  const float* b5a = (const float*)d_in[36], *b5b = (const float*)d_in[37];
  const float* bff = (const float*)d_in[38];
  const float* w1a = (const float*)d_in[39], *w1b = (const float*)d_in[40];
  const float* w2a = (const float*)d_in[41], *w2b = (const float*)d_in[42];
  const float* w3a = (const float*)d_in[43], *w3b = (const float*)d_in[44];
  const float* w4a = (const float*)d_in[45], *w4b = (const float*)d_in[46];
  const float* w5a = (const float*)d_in[47], *w5b = (const float*)d_in[48];
  const float* wf  = (const float*)d_in[49];
  const float* wu1 = (const float*)d_in[50], *wu2 = (const float*)d_in[51];

  // ---- workspace layout (~379 MB linear; cat1/u1/mid overlay scatter grid) ----
  char* ws = (char*)d_ws;
  size_t off = 0;
  auto alloc = [&](size_t bytes) { size_t o = off; off += (bytes + 255) & ~(size_t)255; return o; };
  auto B16 = [&](size_t e) { return (__bf16*)(ws + alloc(e * 2)); };
  auto F32 = [&](size_t e) { return (float*)(ws + alloc(e * 4)); };

  float* cmat = F32(48);
  __bf16 *bw0[5], *bw1[5], *bws[5];
  for (int i = 0; i < 5; ++i) {
    bw0[i] = B16(128 * 256); bw1[i] = B16(128 * 128); bws[i] = B16(128 * 256);
  }
  __bf16* cw1a = B16(9 * 64 * 128);   __bf16* cw1b = B16(9 * 64 * 64);
  __bf16* cw2a = B16(9 * 128 * 64);   __bf16* cw2b = B16(9 * 128 * 128);
  __bf16* cw3a = B16(9 * 256 * 128);  __bf16* cw3b = B16(9 * 256 * 256);
  __bf16* cw4a = B16(9 * 128 * 256);  __bf16* cw4b = B16(9 * 128 * 128);
  __bf16* cw5a = B16(9 * 64 * 128);   __bf16* cw5b = B16(9 * 64 * 64);
  __bf16* cwf  = B16(128 * 64);
  __bf16* cwu2 = B16(4 * 128 * 256);  __bf16* cwu1 = B16(4 * 64 * 128);

  __bf16* X0   = B16(32768ull * 256);
  __bf16* Hb   = B16(32768ull * 128);
  float*  Tmp  = F32(32768ull * 128);
  __bf16* NET  = B16(32768ull * 128);
  __bf16* pool = B16(16 * 128);
  float*  cgrid = F32(33554432ull);            // 16*4*64*64*128, 134 MB
  float*  cnt   = F32(262144ull);              // 16*4*64*64
  __bf16* g0 = B16(64ull * 64 * 64 * 128);
  __bf16* d1 = B16(64ull * 64 * 64 * 64);
  __bf16* p1 = B16(64ull * 32 * 32 * 64);
  __bf16* d2 = B16(64ull * 32 * 32 * 128);
  __bf16* p2 = B16(64ull * 16 * 16 * 128);
  __bf16* d3 = B16(64ull * 16 * 16 * 256);
  __bf16* cat2 = B16(64ull * 32 * 32 * 256);
  __bf16* u2   = B16(64ull * 32 * 32 * 128);
  // overlay (cgrid dead once g0 is built): 67MB + 33.5MB + 33.5MB = 134MB exact
  __bf16* cat1 = (__bf16*)cgrid;
  __bf16* u1   = (__bf16*)((char*)cgrid + 67108864ull);
  __bf16* mid  = (__bf16*)((char*)cgrid + 100663296ull);

  // ---- constants + weight preparation ----
  k_init_cmat<<<1, 32, 0, stream>>>(cmat, (float*)d_out + 33554432ull);
  auto cvt = [&](const float* s, __bf16* d, size_t n) {
    k_cvt_bf16<<<(unsigned)((n + 255) / 256), 256, 0, stream>>>(s, d, n);
  };
  for (int i = 0; i < 5; ++i) {
    cvt(rw0[i], bw0[i], 128 * 256);
    cvt(rw1[i], bw1[i], 128 * 128);
    cvt(rws[i], bws[i], 128 * 256);
  }
  auto pc = [&](const float* s, __bf16* d, int O, int I, int KH, int KW) {
    size_t n = (size_t)O * I * KH * KW;
    k_prep_conv<<<(unsigned)((n + 255) / 256), 256, 0, stream>>>(s, d, O, I, KH, KW);
  };
  pc(w1a, cw1a, 64, 128, 3, 3);  pc(w1b, cw1b, 64, 64, 3, 3);
  pc(w2a, cw2a, 128, 64, 3, 3);  pc(w2b, cw2b, 128, 128, 3, 3);
  pc(w3a, cw3a, 256, 128, 3, 3); pc(w3b, cw3b, 256, 256, 3, 3);
  pc(w4a, cw4a, 128, 256, 3, 3); pc(w4b, cw4b, 128, 128, 3, 3);
  pc(w5a, cw5a, 64, 128, 3, 3);  pc(w5b, cw5b, 64, 64, 3, 3);
  pc(wf, cwf, 128, 64, 1, 1);
  k_prep_up<<<(unsigned)((4ull * 256 * 128 + 255) / 256), 256, 0, stream>>>(wu2, cwu2, 256, 128);
  k_prep_up<<<(unsigned)((4ull * 128 * 64 + 255) / 256), 256, 0, stream>>>(wu1, cwu1, 128, 64);

  // ---- encoder ----
  k_fcpos<<<32768, 256, 0, stream>>>(p, fcw, fcb, X0);   // 32768 rows x 256
  for (int blk = 0; blk < 5; ++blk) {
    if (blk > 0) {
      k_poolT<<<16, 128, 0, stream>>>(NET, pool);
      k_buildX<<<32768, 256, 0, stream>>>(NET, pool, X0);
    }
    // h = relu(x) @ w0^T + b0
    gemm_bf16_wmma<true><<<512, 128, 0, stream>>>(X0, bw0[blk], rb0[blk], nullptr,
                                                  Hb, nullptr, 32768, 128, 256);
    // dx = relu(h) @ w1^T + b1   (f32)
    gemm_bf16_wmma<true><<<512, 128, 0, stream>>>(Hb, bw1[blk], rb1[blk], nullptr,
                                                  nullptr, Tmp, 32768, 128, 128);
    // net = x @ ws^T + dx
    gemm_bf16_wmma<false><<<512, 128, 0, stream>>>(X0, bws[blk], nullptr, Tmp,
                                                   NET, nullptr, 32768, 128, 256);
  }

  // ---- plane scatter + normalization ----
  hipMemsetAsync(cgrid, 0, 33554432ull * 4, stream);
  hipMemsetAsync(cnt, 0, 262144ull * 4, stream);
  k_scatter<<<32768, 128, 0, stream>>>(NET, cmat, p, cgrid, cnt);
  k_norm_g0<<<131072, 256, 0, stream>>>(cgrid, cnt, g0);

  // ---- UNet (all on wmma conv) ----
  conv_nhwc_wmma<3><<<dim3(4096, 1), 128, 0, stream>>>(g0, cw1a, b1a, mid, nullptr, 64, 64, 64, 128, 64, 1);
  conv_nhwc_wmma<3><<<dim3(4096, 1), 128, 0, stream>>>(mid, cw1b, b1b, d1, nullptr, 64, 64, 64, 64, 64, 1);
  k_pool2<<<16384, 256, 0, stream>>>(d1, p1, 64, 64, 64, 64);
  conv_nhwc_wmma<3><<<dim3(1024, 2), 128, 0, stream>>>(p1, cw2a, b2a, mid, nullptr, 64, 32, 32, 64, 128, 1);
  conv_nhwc_wmma<3><<<dim3(1024, 2), 128, 0, stream>>>(mid, cw2b, b2b, d2, nullptr, 64, 32, 32, 128, 128, 1);
  k_pool2<<<8192, 256, 0, stream>>>(d2, p2, 64, 32, 32, 128);
  conv_nhwc_wmma<3><<<dim3(256, 4), 128, 0, stream>>>(p2, cw3a, b3a, mid, nullptr, 64, 16, 16, 128, 256, 1);
  conv_nhwc_wmma<3><<<dim3(256, 4), 128, 0, stream>>>(mid, cw3b, b3b, d3, nullptr, 64, 16, 16, 256, 256, 1);
  upconv_nhwc_wmma<<<dim3(256, 2, 4), 128, 0, stream>>>(d3, cwu2, cat2, 64, 16, 16, 256, 128, 256);
  k_copych<<<32768, 256, 0, stream>>>(d2, cat2, 64 * 32 * 32, 128, 256, 128);
  conv_nhwc_wmma<3><<<dim3(1024, 2), 128, 0, stream>>>(cat2, cw4a, b4a, mid, nullptr, 64, 32, 32, 256, 128, 1);
  conv_nhwc_wmma<3><<<dim3(1024, 2), 128, 0, stream>>>(mid, cw4b, b4b, u2, nullptr, 64, 32, 32, 128, 128, 1);
  upconv_nhwc_wmma<<<dim3(1024, 1, 4), 128, 0, stream>>>(u2, cwu1, cat1, 64, 32, 32, 128, 64, 128);
  k_copych<<<65536, 256, 0, stream>>>(d1, cat1, 64 * 64 * 64, 64, 128, 64);
  conv_nhwc_wmma<3><<<dim3(4096, 1), 128, 0, stream>>>(cat1, cw5a, b5a, mid, nullptr, 64, 64, 64, 128, 64, 1);
  conv_nhwc_wmma<3><<<dim3(4096, 1), 128, 0, stream>>>(mid, cw5b, b5b, u1, nullptr, 64, 64, 64, 64, 64, 1);
  // final 1x1 conv writes f32 directly into d_out (NHWC == reference layout)
  conv_nhwc_wmma<1><<<dim3(4096, 2), 128, 0, stream>>>(u1, cwf, bff, nullptr, (float*)d_out, 64, 64, 64, 64, 128, 0);
}
